// PCNN_ATT_47158740910711
// MI455X (gfx1250) — matrix operand order repaired and checked by reading, not compiled, not run on hardware
//
#include <hip/hip_runtime.h>

// PCNN-ATT fused pipeline for gfx1250 (MI455X), bf16 WMMA path.
//
// Workspace layout (bytes from d_ws):
//   [0)          m_bf    : [64][704] bf16  (m = r_embed @ att_W, padded, row-major)
//   [90112)      remb_bf : [64][704] bf16  (r_embed padded, row-major)
//   [180224)     featB   : [512][128][704] bf16  (features, [s][f] layout)
//   [+92274688)  featT   : [512][704][128] bf16  (features, [f][s] layout)
// total ~184.7 MB workspace.

#define N_REL 53
#define RPAD  64
#define FEAT  690
#define FPAD  704
#define BATCH 512
#define SMAX  128
#define NEGBIG (-1e30f)

typedef __attribute__((ext_vector_type(16))) __bf16 bf16x16;
typedef __attribute__((ext_vector_type(8)))  __bf16 bf16x8;
typedef __attribute__((ext_vector_type(8)))  float  f32x8;

union ABFrag { bf16x16 v; bf16x8 h[2]; };

// A fragment: 16x32 bf16 tile, row-major source with `stride` elements/row.
// Lane L (0-15): row M=L, K=0..15; lanes 16-31: row M=L-16,
// K=16..31  (contiguous order, matching load_B32x16).
__device__ __forceinline__ bf16x16 load_A16x32(const __bf16* base, int stride, int lane) {
  const int m    = lane & 15;
  const int koff = (lane < 16) ? 0 : 16;
  ABFrag f;
  f.v = *(const bf16x16*)(base + m * stride + koff);
  return f.v;
}

// B fragment: 32x16 bf16 tile, sourced from B^T stored row-major ([n][k]),
// `stride` elements per n-row. Lane L (0-15): col n=L, K=0..15; lanes 16-31:
// col n=L-16, K=16..31 -> one contiguous 32B load per lane.
__device__ __forceinline__ bf16x16 load_B32x16(const __bf16* baseT, int stride, int lane) {
  const int n    = lane & 15;
  const int koff = (lane < 16) ? 0 : 16;
  return *(const bf16x16*)(baseT + n * stride + koff);
}

// ---------------------------------------------------------------------------
// Kernel 1: m[r][g] = sum_f r_embed[r,f] * att_W[r,f,g]  (bandwidth on att_W)
// Also packs r_embed to bf16 padded [64][704]. Pads are zero.
// ---------------------------------------------------------------------------
__global__ __launch_bounds__(256)
void pcnn_prep_m(const float* __restrict__ r_embed,
                 const float* __restrict__ att_W,
                 __bf16* __restrict__ m_bf,
                 __bf16* __restrict__ remb_bf) {
  const int r = blockIdx.x;                      // 0..63 (rows >=53 are zero pad)
  const int g = blockIdx.y * 256 + threadIdx.x;  // column, coalesced over att_W
  if (g >= FPAD) return;
  float acc = 0.f, re = 0.f;
  if (r < N_REL && g < FEAT) {
    const float* er = r_embed + (size_t)r * FEAT;
    const float* wr = att_W + (size_t)r * FEAT * FEAT + g;
    for (int f = 0; f < FEAT; ++f) acc += er[f] * wr[(size_t)f * FEAT];
    re = er[g];
  }
  m_bf[r * FPAD + g]    = (__bf16)acc;
  remb_bf[r * FPAD + g] = (__bf16)re;
}

// ---------------------------------------------------------------------------
// Kernel 2: features f32 -> bf16 in two layouts (featB=[s][f], featT=[f][s]).
// Transpose goes through an LDS tile so both reads and writes are coalesced.
// ---------------------------------------------------------------------------
__global__ __launch_bounds__(256)
void pcnn_convert(const float* __restrict__ features,
                  __bf16* __restrict__ featB,
                  __bf16* __restrict__ featT) {
  __shared__ float tile[32][33];
  const int b   = blockIdx.x;
  const int tid = threadIdx.x;
  const float* fb = features + (size_t)b * SMAX * FEAT;
  __bf16* oB = featB + (size_t)b * SMAX * FPAD;
  __bf16* oT = featT + (size_t)b * FPAD * SMAX;

  // [s][f] layout, zero-padded f in [690,704)
  for (int idx = tid; idx < SMAX * FPAD; idx += 256) {
    const int s = idx / FPAD, g = idx - s * FPAD;
    const float v = (g < FEAT) ? fb[s * FEAT + g] : 0.f;
    oB[idx] = (__bf16)v;
  }

  // [f][s] layout via 32x32 LDS transpose tiles
  const int tx = tid & 31, ty = tid >> 5;
  for (int sb = 0; sb < SMAX / 32; ++sb) {
    for (int fbk = 0; fbk < FPAD / 32; ++fbk) {
      __syncthreads();
      for (int yy = ty; yy < 32; yy += 8) {
        const int s = sb * 32 + yy, f = fbk * 32 + tx;
        tile[yy][tx] = (f < FEAT) ? fb[s * FEAT + f] : 0.f;
      }
      __syncthreads();
      for (int yy = ty; yy < 32; yy += 8) {
        const int f = fbk * 32 + yy, s = sb * 32 + tx;
        oT[f * SMAX + s] = (__bf16)tile[tx][yy];
      }
    }
  }
}

// ---------------------------------------------------------------------------
// Kernel 3: fused per-bag attention. One block (8 wave32) per bag.
//   GEMM1: logits[16x128] = m[16xK] * feat^T[Kx128]   (K=704, 22 wmma steps)
//   softmax over valid sentences (length mask)
//   GEMM2: sfeat[16x704] = w[16x128] * feat[128x704]  (x0.5, -> bf16 LDS)
//   GEMM3: scores[16x64] = sfeat[16xK] * r_embed^T[Kx64]
//   epilogue: +bias, log_softmax over k, max over r -> out[b][53]
// ---------------------------------------------------------------------------
__global__ __launch_bounds__(256)
void pcnn_att_fused(const __bf16* __restrict__ m_bf,
                    const __bf16* __restrict__ remb_bf,
                    const __bf16* __restrict__ featB,
                    const __bf16* __restrict__ featT,
                    const float*  __restrict__ r_bias,
                    const int*    __restrict__ lengths,
                    float* __restrict__ out) {
  __shared__ __align__(32) float  s_logits[16 * SMAX];   //  8 KB
  __shared__ __align__(32) __bf16 s_w[16 * SMAX];        //  4 KB
  __shared__ __align__(32) __bf16 s_sfeat[16 * FPAD];    // 22 KB
  __shared__ __align__(32) float  s_scores[RPAD * RPAD]; // 16 KB
  __shared__ float s_lse[RPAD];

  const int b    = blockIdx.x;
  const int tid  = threadIdx.x;
  const int wv   = tid >> 5;
  const int lane = tid & 31;
  const int len  = lengths[b];
  const int hi8  = (lane < 16) ? 0 : 8;
  const int n15  = lane & 15;

  const __bf16* featB_b = featB + (size_t)b * SMAX * FPAD;
  const __bf16* featT_b = featT + (size_t)b * FPAD * SMAX;

  for (int rt = 0; rt < 4; ++rt) {
    const int r0 = rt * 16;

    // ---- GEMM1: each wave owns one 16-wide s-tile -------------------------
    {
      const int st = wv;  // 8 waves <-> 8 s-tiles of 128
      f32x8 acc = {};
      for (int ks = 0; ks < FPAD / 32; ++ks) {
        __builtin_prefetch(featB_b + (st * 16) * FPAD + (ks + 1) * 32, 0, 1);
        bf16x16 a  = load_A16x32(m_bf + r0 * FPAD + ks * 32, FPAD, lane);
        bf16x16 bm = load_B32x16(featB_b + (st * 16) * FPAD + ks * 32, FPAD, lane);
        acc = __builtin_amdgcn_wmma_f32_16x16x32_bf16(false, a, false, bm,
                                                      (short)0, acc, false, false);
      }
#pragma unroll
      for (int i = 0; i < 8; ++i)
        s_logits[(i + hi8) * SMAX + st * 16 + n15] = acc[i];
    }
    __syncthreads();

    // ---- masked softmax over sentences (one thread per relation row) ------
    if (tid < 16) {
      float* row = s_logits + tid * SMAX;
      float mx = NEGBIG;
      for (int s = 0; s < len; ++s) mx = fmaxf(mx, row[s]);
      float sum = 0.f;
      for (int s = 0; s < SMAX; ++s) {
        const float e = (s < len) ? __expf(row[s] - mx) : 0.f;
        row[s] = e;
        sum += e;
      }
      const float inv = 1.f / sum;
      for (int s = 0; s < SMAX; ++s)
        s_w[tid * SMAX + s] = (__bf16)(row[s] * inv);
    }
    __syncthreads();

    // ---- GEMM2: weighted sentence sum, scaled by 0.5 ----------------------
    for (int ft = wv; ft < FPAD / 16; ft += 8) {
      f32x8 acc = {};
      for (int ks = 0; ks < SMAX / 32; ++ks) {
        bf16x16 a  = load_A16x32(s_w + ks * 32, SMAX, lane);
        bf16x16 bm = load_B32x16(featT_b + (ft * 16) * SMAX + ks * 32, SMAX, lane);
        acc = __builtin_amdgcn_wmma_f32_16x16x32_bf16(false, a, false, bm,
                                                      (short)0, acc, false, false);
      }
#pragma unroll
      for (int i = 0; i < 8; ++i)
        s_sfeat[(i + hi8) * FPAD + ft * 16 + n15] = (__bf16)(acc[i] * 0.5f);
    }
    __syncthreads();

    // ---- GEMM3: relation scores for this r-tile ---------------------------
    if (wv < 4) {
      const int nt = wv;  // 4 n-tiles of 64 relation columns
      f32x8 acc = {};
      for (int ks = 0; ks < FPAD / 32; ++ks) {
        bf16x16 a  = load_A16x32(s_sfeat + ks * 32, FPAD, lane);
        bf16x16 bm = load_B32x16(remb_bf + (nt * 16) * FPAD + ks * 32, FPAD, lane);
        acc = __builtin_amdgcn_wmma_f32_16x16x32_bf16(false, a, false, bm,
                                                      (short)0, acc, false, false);
      }
#pragma unroll
      for (int i = 0; i < 8; ++i)
        s_scores[(r0 + i + hi8) * RPAD + nt * 16 + n15] = acc[i];
    }
    __syncthreads();
  }

  // ---- epilogue: bias + log_softmax over k, then max over r ---------------
  if (tid < N_REL) {
    const float* row = s_scores + tid * RPAD;
    float mx = NEGBIG;
    for (int k = 0; k < N_REL; ++k) mx = fmaxf(mx, row[k] + r_bias[k]);
    float sum = 0.f;
    for (int k = 0; k < N_REL; ++k) sum += __expf(row[k] + r_bias[k] - mx);
    s_lse[tid] = mx + __logf(sum);
  }
  __syncthreads();
  if (tid < N_REL) {
    const int k = tid;
    float best = NEGBIG;
    for (int r = 0; r < N_REL; ++r)
      best = fmaxf(best, s_scores[r * RPAD + k] + r_bias[k] - s_lse[r]);
    out[b * N_REL + k] = best;
  }
}

// ---------------------------------------------------------------------------
extern "C" void kernel_launch(void* const* d_in, const int* in_sizes, int n_in,
                              void* d_out, int out_size, void* d_ws, size_t ws_size,
                              hipStream_t stream) {
  (void)in_sizes; (void)n_in; (void)out_size; (void)ws_size;
  const float* features = (const float*)d_in[0];  // [512,128,690] f32
  const int*   lengths  = (const int*)d_in[1];    // [512] i32
  const float* r_embed  = (const float*)d_in[2];  // [53,690] f32
  const float* r_bias   = (const float*)d_in[3];  // [53] f32
  const float* att_W    = (const float*)d_in[4];  // [53,690,690] f32
  float* out = (float*)d_out;                     // [512,53] f32

  char* ws = (char*)d_ws;
  __bf16* m_bf  = (__bf16*)(ws);
  __bf16* remb  = (__bf16*)(ws + 90112);
  __bf16* featB = (__bf16*)(ws + 180224);
  __bf16* featT = (__bf16*)(ws + 180224 + (size_t)BATCH * SMAX * FPAD * 2);

  pcnn_prep_m<<<dim3(RPAD, 3), 256, 0, stream>>>(r_embed, att_W, m_bf, remb);
  pcnn_convert<<<BATCH, 256, 0, stream>>>(features, featB, featT);
  pcnn_att_fused<<<BATCH, 256, 0, stream>>>(m_bf, remb, featB, featT,
                                            r_bias, lengths, out);
}
